// MBLFE_90812788507332
// MI455X (gfx1250) — matrix operand order repaired and checked by reading, not compiled, not run on hardware
//
#include <hip/hip_runtime.h>
#include <hip/hip_bf16.h>
#include <math.h>

// Problem constants (match reference)
#define B_ 16384
#define D_ 512
#define E_ 16
#define H_ 256
#define L_ 64

typedef __bf16 bf16;
typedef __attribute__((ext_vector_type(8)))  __bf16 bf16x8;
typedef __attribute__((ext_vector_type(16))) __bf16 bf16x16;
typedef __attribute__((ext_vector_type(8)))  float  floatx8;

// -------------------------------------------------------------------------
// Fragment helper: build a 16-element bf16 A/B fragment from two contiguous
// 16-byte chunks (elements 0..7 from p0, 8..15 from p1).
// -------------------------------------------------------------------------
__device__ __forceinline__ bf16x16 make_frag(const bf16* p0, const bf16* p1) {
  bf16x8 lo = *(const bf16x8*)p0;
  bf16x8 hi = *(const bf16x8*)p1;
  bf16x16 f;
#pragma unroll
  for (int i = 0; i < 8; ++i) { f[i] = lo[i]; f[i + 8] = hi[i]; }
  return f;
}

// -------------------------------------------------------------------------
// Branch-free tanh. Prefer CDNA5's hardware V_TANH_F32 (TRANS32 pipe, can
// co-execute with WMMA); otherwise a divergence-free exp2-based identity.
// -------------------------------------------------------------------------
__device__ __forceinline__ float fast_tanh(float v) {
#if __has_builtin(__builtin_amdgcn_tanhf)
  return __builtin_amdgcn_tanhf(v);
#else
  float c = fminf(fmaxf(v, -15.0f), 15.0f);   // keep exp finite
  float t = __expf(2.0f * c);                 // v_exp_f32, no branches
  return (t - 1.0f) / (t + 1.0f);
#endif
}

// -------------------------------------------------------------------------
// Gating kernel: one thread per (b, e). Softmax/threshold across the 16
// experts lives within a 16-lane group (wave32 -> two rows per wave).
// -------------------------------------------------------------------------
__global__ __launch_bounds__(256) void moe_gating_kernel(
    const float* __restrict__ x,       // [B, D]
    const float* __restrict__ noise,   // [B, E]
    const float* __restrict__ wg,      // [D, E]
    const float* __restrict__ wn,      // [D, E]
    float* __restrict__ gates)         // [B, E]
{
  const int tid = blockIdx.x * 256 + threadIdx.x;
  const int b = tid >> 4;
  const int e = tid & 15;

  const float* xr = x + (size_t)b * D_;
  float accg = 0.0f, accn = 0.0f;
  for (int j = 0; j < D_; ++j) {
    float xv = xr[j];
    accg = fmaf(xv, wg[j * E_ + e], accg);
    accn = fmaf(xv, wn[j * E_ + e], accn);
  }

  // softplus(accn) + eps
  float sp = (accn > 15.0f) ? accn : log1pf(expf(accn));
  float stddev = sp + 0.01f;
  float logit = accg + noise[(size_t)b * E_ + e] * stddev;

  // softmax over the 16 experts (16-lane shuffle group)
  float m = logit;
#pragma unroll
  for (int o = 8; o >= 1; o >>= 1) m = fmaxf(m, __shfl_xor(m, o, 16));
  float ex = expf(logit - m);
  float s = ex;
#pragma unroll
  for (int o = 8; o >= 1; o >>= 1) s += __shfl_xor(s, o, 16);
  float p = ex / s;

  // mean over experts (== 1/16 analytically, computed for fidelity)
  float ps = p;
#pragma unroll
  for (int o = 8; o >= 1; o >>= 1) ps += __shfl_xor(ps, o, 16);
  float mean = ps * (1.0f / 16.0f) - 1e-8f;

  float g = (p >= mean) ? p : 0.0f;
  float gs = g;
#pragma unroll
  for (int o = 8; o >= 1; o >>= 1) gs += __shfl_xor(gs, o, 16);
  gates[(size_t)b * E_ + e] = g / gs;
}

// -------------------------------------------------------------------------
// fp32 -> bf16 conversion (x)
// -------------------------------------------------------------------------
__global__ __launch_bounds__(256) void moe_cvt_x_kernel(
    const float* __restrict__ x, bf16* __restrict__ xbf, int n)
{
  int i = blockIdx.x * 256 + threadIdx.x;
  if (i < n) xbf[i] = (bf16)x[i];
}

// W1 [E, D, H] -> W1t bf16 [E, H, D]
__global__ __launch_bounds__(256) void moe_tw1_kernel(
    const float* __restrict__ W1, bf16* __restrict__ w1t)
{
  int i = blockIdx.x * 256 + threadIdx.x;   // over E*H*D
  int d = i & (D_ - 1);
  int r = i >> 9;
  int h = r & (H_ - 1);
  int e = r >> 8;
  w1t[i] = (bf16)W1[((size_t)e * D_ + d) * H_ + h];
}

// W2 [E, H, L] -> W2t bf16 [E, L, H]
__global__ __launch_bounds__(256) void moe_tw2_kernel(
    const float* __restrict__ W2, bf16* __restrict__ w2t)
{
  int i = blockIdx.x * 256 + threadIdx.x;   // over E*L*H
  int h = i & (H_ - 1);
  int r = i >> 8;
  int l = r & (L_ - 1);
  int e = r >> 6;
  w2t[i] = (bf16)W2[((size_t)e * H_ + h) * L_ + l];
}

// -------------------------------------------------------------------------
// Fused expert MLP: one wave per (16-row tile, expert).
//   GEMM1: h = tanh(x[16,512] @ W1[e]  + b1)   via v_wmma_f32_16x16x32_bf16
//   GEMM2: y = h[16,256]   @ W2[e]  + b2       via v_wmma_f32_16x16x32_bf16
//   out   = gates[:,e,None] * y
// A-fragment per ISA 16-bit layout: lanes 0-15 K=+0..7/+16..23,
// lanes 16-31 K=+8..15/+24..31. B-fragment (from transposed weights):
// lanes 0-15 hold K=k0+0..7/+16..23 of column n, lanes 16-31 K=k0+8..15/+24..31.
// -------------------------------------------------------------------------
__global__ __launch_bounds__(128) void moe_mlp_kernel(
    const bf16*  __restrict__ xbf,    // [B, D]
    const bf16*  __restrict__ w1t,    // [E, H, D]
    const bf16*  __restrict__ w2t,    // [E, L, H]
    const float* __restrict__ b1,     // [E, H]
    const float* __restrict__ b2,     // [E, L]
    const float* __restrict__ gates,  // [B, E]
    float*       __restrict__ out)    // [B, E, L]
{
  __shared__ __align__(16) bf16 lds_h[4][16][H_ + 8];  // 33,792 B

  const int wave = threadIdx.x >> 5;
  const int lane = threadIdx.x & 31;
  const int hf   = lane >> 4;     // 0 or 1 (half-wave)
  const int lr   = lane & 15;     // row (A/M) or column (B/N, C/N) index
  const int tile = blockIdx.x;            // 0..1023 : 16-row tile of x
  const int e    = blockIdx.y * 4 + wave; // 0..15   : expert
  const int row0 = tile * 16;

  const bf16* xrow = xbf + (size_t)(row0 + lr) * D_;
  const bf16* w1e  = w1t + (size_t)e * H_ * D_;
  const bf16* w2e  = w2t + (size_t)e * L_ * H_;

  // ---------------- GEMM1: 4 chunks of 64 H-columns each ----------------
  for (int nc = 0; nc < 4; ++nc) {
    floatx8 acc[4];
#pragma unroll
    for (int t = 0; t < 4; ++t) {
      float bv = b1[e * H_ + nc * 64 + t * 16 + lr];
#pragma unroll
      for (int r = 0; r < 8; ++r) acc[t][r] = bv;
    }

    for (int k0 = 0; k0 < D_; k0 += 32) {
      const int ka = k0 + hf * 8;
      bf16x16 a = make_frag(xrow + ka, xrow + ka + 16);
      const int kb = k0 + hf * 8;
      // Prefetch next K-chunk of this wave's private weight stream
      // (lowers to global_prefetch_b8; speculative, safe at row ends).
      __builtin_prefetch(w1e + (size_t)(nc * 64 + lr) * D_ + kb + 32, 0, 1);
#pragma unroll
      for (int t = 0; t < 4; ++t) {
        const bf16* bp = w1e + (size_t)(nc * 64 + t * 16 + lr) * D_ + kb;
        bf16x16 bm = make_frag(bp, bp + 16);
        acc[t] = __builtin_amdgcn_wmma_f32_16x16x32_bf16(
            false, a, false, bm, (short)0, acc[t], false, false);
      }
    }

    // tanh + bf16 convert, stash h tile in this wave's LDS slice
#pragma unroll
    for (int t = 0; t < 4; ++t) {
#pragma unroll
      for (int r = 0; r < 8; ++r) {
        float v = fast_tanh(acc[t][r]);
        lds_h[wave][r + 8 * hf][nc * 64 + t * 16 + lr] = (bf16)v;
      }
    }
  }

  // ---------------- GEMM2: y[16,64] = h @ W2[e] + b2 ----------------
  floatx8 yacc[4];
#pragma unroll
  for (int t = 0; t < 4; ++t) {
    float bv = b2[e * L_ + t * 16 + lr];
#pragma unroll
    for (int r = 0; r < 8; ++r) yacc[t][r] = bv;
  }

  for (int k0 = 0; k0 < H_; k0 += 32) {
    const int ka = k0 + hf * 8;
    const bf16* hp = &lds_h[wave][lr][ka];
    bf16x16 a = make_frag(hp, hp + 16);
    const int kb = k0 + hf * 8;
#pragma unroll
    for (int t = 0; t < 4; ++t) {
      const bf16* bp = w2e + (size_t)(t * 16 + lr) * H_ + kb;
      bf16x16 bm = make_frag(bp, bp + 16);
      yacc[t] = __builtin_amdgcn_wmma_f32_16x16x32_bf16(
          false, a, false, bm, (short)0, yacc[t], false, false);
    }
  }

  // ---------------- scale by gates, store out[b, e, l] ----------------
#pragma unroll
  for (int r = 0; r < 8; ++r) {
    const int m = r + 8 * hf;               // C layout: lanes16-31 -> M=8+r
    const int brow = row0 + m;
    const float g = gates[(size_t)brow * E_ + e];
#pragma unroll
    for (int t = 0; t < 4; ++t) {
      out[((size_t)brow * E_ + e) * L_ + t * 16 + lr] = g * yacc[t][r];
    }
  }
}

// -------------------------------------------------------------------------
// Host-side launcher
// -------------------------------------------------------------------------
extern "C" void kernel_launch(void* const* d_in, const int* in_sizes, int n_in,
                              void* d_out, int out_size, void* d_ws, size_t ws_size,
                              hipStream_t stream) {
  const float* x      = (const float*)d_in[0];  // [B, D]
  const float* noise  = (const float*)d_in[1];  // [B, E]
  const float* w_gate = (const float*)d_in[2];  // [D, E]
  const float* w_noise= (const float*)d_in[3];  // [D, E]
  const float* W1     = (const float*)d_in[4];  // [E, D, H]
  const float* b1     = (const float*)d_in[5];  // [E, H]
  const float* W2     = (const float*)d_in[6];  // [E, H, L]
  const float* b2     = (const float*)d_in[7];  // [E, L]
  float* out = (float*)d_out;                   // [B, E, L]

  // Workspace layout (all 16B aligned):
  //   gates f32 [B,E]        : 1,048,576 B
  //   xbf  bf16 [B,D]        : 16,777,216 B
  //   w1t  bf16 [E,H,D]      : 4,194,304 B
  //   w2t  bf16 [E,L,H]      : 524,288 B   (total ~22.5 MB)
  char* ws = (char*)d_ws;
  float* gates = (float*)ws;
  bf16*  xbf   = (bf16*)(ws + (size_t)1048576);
  bf16*  w1t   = (bf16*)(ws + (size_t)1048576 + 16777216);
  bf16*  w2t   = (bf16*)(ws + (size_t)1048576 + 16777216 + 4194304);

  // 1) gating (B*E threads)
  moe_gating_kernel<<<dim3((B_ * E_) / 256), dim3(256), 0, stream>>>(
      x, noise, w_gate, w_noise, gates);

  // 2) conversions / transposes to bf16
  moe_cvt_x_kernel<<<dim3((B_ * D_) / 256), dim3(256), 0, stream>>>(
      x, xbf, B_ * D_);
  moe_tw1_kernel<<<dim3((E_ * H_ * D_) / 256), dim3(256), 0, stream>>>(W1, w1t);
  moe_tw2_kernel<<<dim3((E_ * L_ * H_) / 256), dim3(256), 0, stream>>>(W2, w2t);

  // 3) fused expert MLP: grid (row-tiles, expert-groups), 4 waves per block
  moe_mlp_kernel<<<dim3(B_ / 16, E_ / 4), dim3(128), 0, stream>>>(
      xbf, w1t, w2t, b1, b2, gates, out);
}
